// NODE_RNN_32349693674174
// MI455X (gfx1250) — matrix operand order repaired and checked by reading, not compiled, not run on hardware
//
#include <hip/hip_runtime.h>

// Problem constants (from reference): B=16, T=512, D=32, H=128, K=8, C=2
#define BB 16
#define TT 512
#define DD 32
#define HH 128
#define KB 8
#define I_ODE 160      // H + D
#define KS_ODE 40      // 1280 / 32
#define KSH 32         // h-based k-steps (i < 128)
#define KSPH 4         // h-based k-steps per wave (32 / 8)
#define I_CI  128
#define KS_CI 32       // 1024 / 32
#define KSP_CI 4
#define NWAVE 8
#define HP (HH + 1)    // padded LDS row stride (129 words -> conflict-free)

typedef _Float16 v16h __attribute__((ext_vector_type(16)));
typedef float    v8f  __attribute__((ext_vector_type(8)));

__device__ __forceinline__ float ftanh(float x) {
#if __has_builtin(__builtin_amdgcn_tanhf)
    return __builtin_amdgcn_tanhf(x);      // v_tanh_f32 on gfx1250
#else
    float e = __expf(2.0f * x);
    return 1.0f - 2.0f / (e + 1.0f);
#endif
}

// Build one 32B A-fragment slot: halves j=0..7 -> tanh(v0*s[kb]+ncs[kb]),
// j=8..15 -> same with v1.  (tanh((v-c)*s) with ncs = -c*s precomputed)
__device__ __forceinline__ v16h build_frag(float v0, float v1,
                                           const float* __restrict__ s,
                                           const float* __restrict__ ncs) {
    v16h f;
    #pragma unroll
    for (int kb = 0; kb < KB; ++kb) {
        f[kb]     = (_Float16)ftanh(fmaf(v0, s[kb], ncs[kb]));
        f[kb + 8] = (_Float16)ftanh(fmaf(v1, s[kb], ncs[kb]));
    }
    return f;
}

// Pack coeffs (O, I, KB) f32 -> f16 in WMMA B-fragment order.
// B layout (16-bit, 32x16 KxN): lanes 0-15 hold K=0..7 (halves j=0..7), K=16..23 (j=8..15),
// lanes 16-31 hold K=8..15 (j=0..7), K=24..31 (j=8..15), column N = lane&15.
// flat idx = ((w*KS + ks)*32 + lane)*16 + j
__global__ void pack_coeffs(const float* __restrict__ src, _Float16* __restrict__ dst,
                            int I, int KS) {
    int idx = blockIdx.x * blockDim.x + threadIdx.x;
    int total = NWAVE * KS * 32 * 16;
    if (idx >= total) return;
    int j    = idx & 15;
    int lane = (idx >> 4) & 31;
    int ks   = (idx >> 9) % KS;
    int w    = (idx >> 9) / KS;
    int n    = w * 16 + (lane & 15);
    int g    = ks * 32 + (j & 7) + ((lane >> 4) << 3) + ((j >> 3) << 4);   // flattened K index = i*8 + kb
    int i    = g >> 3, kb = g & 7;
    dst[idx] = (_Float16)src[((size_t)n * I + i) * KB + kb];
}

__global__ __launch_bounds__(256, 1)
void node_rnn_kernel(const float* __restrict__ x,
                     const float* __restrict__ lift_w, const float* __restrict__ lift_b,
                     const float* __restrict__ ode_centers, const float* __restrict__ ode_scales,
                     const float* __restrict__ gain, const float* __restrict__ bias,
                     const float* __restrict__ ci_centers, const float* __restrict__ ci_scales,
                     const float* __restrict__ head_w, const float* __restrict__ head_b,
                     const _Float16* __restrict__ bOde, const _Float16* __restrict__ bCi,
                     float* __restrict__ out) {
    // A-fragment staging in WMMA layout: aFrag[ks][lane][j], 32B contiguous per lane.
    __shared__ __align__(32) _Float16 aFrag[KS_ODE][32][16];   // 40 KB
    __shared__ float hstage[BB][HP];                           // padded rows
    __shared__ float cS[4][KB];  // [0]=ode scale  [1]=ode -c*s  [2]=ci scale  [3]=ci -c*s

    const int tid    = threadIdx.x;
    const int wave   = tid >> 5;
    const int lane   = tid & 31;
    const int laneN  = lane & 15;
    const int laneHi = lane >> 4;
    const float dt   = 1.0f / (float)(TT - 1);
    const float hsix = dt / 6.0f;

    if (tid < 8)       cS[0][tid]      = ode_scales[tid];
    else if (tid < 16) cS[1][tid - 8]  = -ode_centers[tid - 8] * ode_scales[tid - 8];
    else if (tid < 24) cS[2][tid - 16] = ci_scales[tid - 16];
    else if (tid < 32) cS[3][tid - 24] = -ci_centers[tid - 24] * ci_scales[tid - 24];

    // h0 = lift_w @ x[:,0,:] + lift_b
    for (int idx = tid; idx < BB * HH; idx += 256) {
        int m = idx >> 7, n = idx & 127;
        float s = lift_b[n];
        const float* xr = x + (size_t)m * TT * DD;
        #pragma unroll 8
        for (int d = 0; d < DD; ++d) s += lift_w[n * DD + d] * xr[d];
        hstage[m][n] = s;
    }
    __syncthreads();

    // Register ownership matches f32 C/D fragment layout:
    //   VGPR r, lane -> (M = r + 8*laneHi, N = wave*16 + laneN)
    const int nOwn = wave * 16 + laneN;
    const float g_n = gain[nOwn];
    const float b_n = bias[nOwn];
    float hreg[8], ksum[8];
    #pragma unroll
    for (int r = 0; r < 8; ++r) { hreg[r] = hstage[r + 8 * laneHi][nOwn]; ksum[r] = 0.0f; }

    // Producer: thread owns fragment lane-slot `lane`.
    //   h-based: ks = wave*4 + c (c=0..3), i0 = ks*4 + laneHi, i1 = i0+2 (< 128)
    //   x-based: ks = 32 + wave,  d0 = wave*4 + laneHi, d1 = d0+2 (< 32)
    const int mP    = laneN;
    const int ksX   = KSH + wave;
    const int d0    = wave * 4 + laneHi;
    const int d1    = d0 + 2;
    const float* xrow = x + (size_t)mP * TT * DD;

    // Preamble: x-based fragments for x(0) (used by stage 0 of t=0).
    *(v16h*)(&aFrag[ksX][lane][0]) = build_frag(xrow[d0], xrow[d1], cS[0], cS[1]);

    for (int t = 0; t < TT - 1; ++t) {
        const float* x0 = xrow + (size_t)t * DD;
        const float* x1 = x0 + DD;
        #pragma unroll
        for (int stage = 0; stage < 4; ++stage) {
            // ---- produce h-based tanh fragments (always) ----
            #pragma unroll
            for (int c = 0; c < KSPH; ++c) {
                const int ks = wave * KSPH + c;
                const int i0 = ks * 4 + laneHi;
                *(v16h*)(&aFrag[ks][lane][0]) =
                    build_frag(hstage[mP][i0], hstage[mP][i0 + 2], cS[0], cS[1]);
            }
            // ---- x-based fragments: only when the x input changes ----
            // stage0 reuses x1-frags of previous step (or preamble);
            // stage2 reuses stage1's xmid-frags.
            if (stage == 1) {
                *(v16h*)(&aFrag[ksX][lane][0]) =
                    build_frag(0.5f * (x0[d0] + x1[d0]),
                               0.5f * (x0[d1] + x1[d1]), cS[0], cS[1]);
            } else if (stage == 3) {
                *(v16h*)(&aFrag[ksX][lane][0]) =
                    build_frag(x1[d0], x1[d1], cS[0], cS[1]);
            }
            __syncthreads();

            // ---- GEMM: phi(16x16 tile) = A(16x1280) x B(1280x16) ----
            v8f acc = {};
            const _Float16* bp = bOde + (((size_t)wave * KS_ODE) * 32 + lane) * 16;
            #pragma unroll 8
            for (int ks = 0; ks < KS_ODE; ++ks) {
                v16h a  = *(const v16h*)(&aFrag[ks][lane][0]);
                v16h bm = *(const v16h*)(bp + (size_t)ks * 512);
                acc = __builtin_amdgcn_wmma_f32_16x16x32_f16(
                        false, a, false, bm, (short)0, acc, false, false);
            }

            // ---- RK4 stage update (stage is compile-time after unroll) ----
            #pragma unroll
            for (int r = 0; r < 8; ++r) {
                float k = ftanh(acc[r]) * g_n + b_n;
                int m = r + 8 * laneHi;
                float vnext;
                if (stage == 0)      { ksum[r] = k;            vnext = hreg[r] + 0.5f * dt * k; }
                else if (stage == 1) { ksum[r] += 2.0f * k;    vnext = hreg[r] + 0.5f * dt * k; }
                else if (stage == 2) { ksum[r] += 2.0f * k;    vnext = hreg[r] + dt * k; }
                else                 { hreg[r] += hsix * (ksum[r] + k); vnext = hreg[r]; }
                hstage[m][nOwn] = vnext;
            }
            __syncthreads();   // hstage visible; safe to overwrite aFrag next stage
        }
    }

    // ---- final KAN cell: hfin = tanh(ci_coeffs . tanh_basis(hT)) ----
    #pragma unroll
    for (int c = 0; c < KSP_CI; ++c) {
        const int ks = wave * KSP_CI + c;
        const int i0 = ks * 4 + laneHi;
        *(v16h*)(&aFrag[ks][lane][0]) =
            build_frag(hstage[mP][i0], hstage[mP][i0 + 2], cS[2], cS[3]);
    }
    __syncthreads();

    {
        v8f acc = {};
        const _Float16* bp = bCi + (((size_t)wave * KS_CI) * 32 + lane) * 16;
        #pragma unroll 8
        for (int ks = 0; ks < KS_CI; ++ks) {
            v16h a  = *(const v16h*)(&aFrag[ks][lane][0]);
            v16h bm = *(const v16h*)(bp + (size_t)ks * 512);
            acc = __builtin_amdgcn_wmma_f32_16x16x32_f16(
                    false, a, false, bm, (short)0, acc, false, false);
        }
        #pragma unroll
        for (int r = 0; r < 8; ++r)
            hstage[r + 8 * laneHi][nOwn] = ftanh(acc[r]);
    }
    __syncthreads();

    // ---- head: out[b,c] = hfin[b,:] . head_w[c,:] + head_b[c] ----
    if (tid < BB * 2) {
        int b = tid >> 1, c = tid & 1;
        float s = head_b[c];
        #pragma unroll 8
        for (int n = 0; n < HH; ++n) s += hstage[b][n] * head_w[c * HH + n];
        out[b * 2 + c] = s;
    }
}

extern "C" void kernel_launch(void* const* d_in, const int* in_sizes, int n_in,
                              void* d_out, int out_size, void* d_ws, size_t ws_size,
                              hipStream_t stream) {
    const float* x           = (const float*)d_in[0];
    const float* lift_w      = (const float*)d_in[1];
    const float* lift_b      = (const float*)d_in[2];
    const float* ode_coeffs  = (const float*)d_in[3];
    const float* ode_centers = (const float*)d_in[4];
    const float* ode_scales  = (const float*)d_in[5];
    const float* gain        = (const float*)d_in[6];
    const float* bias        = (const float*)d_in[7];
    const float* ci_coeffs   = (const float*)d_in[8];
    const float* ci_centers  = (const float*)d_in[9];
    const float* ci_scales   = (const float*)d_in[10];
    // d_in[11..13] = ch_* : dead in the reference (comb[:H] keeps only x_phi)
    const float* head_w      = (const float*)d_in[14];
    const float* head_b      = (const float*)d_in[15];

    const size_t nOde = (size_t)NWAVE * KS_ODE * 32 * 16;   // 163840 halves
    const size_t nCi  = (size_t)NWAVE * KS_CI  * 32 * 16;   // 131072 halves
    _Float16* bOde = (_Float16*)d_ws;
    _Float16* bCi  = (_Float16*)((char*)d_ws + nOde * sizeof(_Float16));

    pack_coeffs<<<(int)((nOde + 255) / 256), 256, 0, stream>>>(ode_coeffs, bOde, I_ODE, KS_ODE);
    pack_coeffs<<<(int)((nCi  + 255) / 256), 256, 0, stream>>>(ci_coeffs,  bCi,  I_CI,  KS_CI);

    node_rnn_kernel<<<1, 256, 0, stream>>>(x, lift_w, lift_b, ode_centers, ode_scales,
                                           gain, bias, ci_centers, ci_scales,
                                           head_w, head_b, bOde, bCi, (float*)d_out);
}
